// GCNNet_4810363372847
// MI455X (gfx1250) — matrix operand order repaired and checked by reading, not compiled, not run on hardware
//
#include <hip/hip_runtime.h>

// ---------- types for WMMA ----------
typedef __attribute__((ext_vector_type(16))) __bf16          v16bf;
typedef __attribute__((ext_vector_type(16))) unsigned short  v16u;
typedef __attribute__((ext_vector_type(8)))  unsigned short  v8u;
typedef __attribute__((ext_vector_type(8)))  float           v8f;

// f32 -> bf16 round-to-nearest-even (bit level)
static __device__ __forceinline__ unsigned short f2bf(float f) {
  unsigned u = __builtin_bit_cast(unsigned, f);
  unsigned r = 0x7FFFu + ((u >> 16) & 1u);
  return (unsigned short)((u + r) >> 16);
}

// ---------- degree / normalization ----------
__global__ void deg_init_k(float* __restrict__ dinv, int n) {
  int i = blockIdx.x * blockDim.x + threadIdx.x;
  if (i < n) dinv[i] = 1.0f;               // self loop contributes 1
}

__global__ void deg_count_k(const int* __restrict__ dst, float* __restrict__ dinv, int E) {
  int e = blockIdx.x * blockDim.x + threadIdx.x;
  if (e < E) atomicAdd(&dinv[dst[e]], 1.0f);
}

__global__ void deg_finish_k(float* __restrict__ dinv, int n) {
  int i = blockIdx.x * blockDim.x + threadIdx.x;
  if (i < n) dinv[i] = rsqrtf(dinv[i]);    // deg >= 1 always
}

// ---------- W pre-transpose + f32->bf16 + zero-pad:  Wt[N][KP] = W[K][N]^T ----------
__global__ void wcvt_k(const float* __restrict__ W, unsigned short* __restrict__ Wt,
                       int K, int KP, int N) {
  int t = blockIdx.x * blockDim.x + threadIdx.x;
  if (t >= N * KP) return;
  int ncol = t / KP;
  int k    = t % KP;
  Wt[t] = (k < K) ? f2bf(W[(long)k * N + ncol]) : (unsigned short)0;
}

// ---------- tiled WMMA GEMM:  Y[n x N] = X[n x K] * W[K x N]  ----------
// One block per 16-row M tile; N/16 waves, one 16-col N tile per wave.
// A staged in LDS (bf16, zero-padded to KP); B read from pre-transposed bf16 Wt.
template <int K, int KP, int N>
__global__ __launch_bounds__((N / 16) * 32)
void gcn_gemm_bf16(const float* __restrict__ X, const unsigned short* __restrict__ Wt,
                   float* __restrict__ Y, int n) {
  __shared__ __align__(32) unsigned short As[16][KP];
  constexpr int NT = (N / 16) * 32;
  const int tid = threadIdx.x;
  const int m0  = blockIdx.x * 16;

  // stage A tile: float4 global loads -> ushort4 DS stores (K and KP divisible by 4)
  constexpr int Q = KP / 4;
  for (int idx = tid; idx < 16 * Q; idx += NT) {
    int row = idx / Q, q = idx % Q;
    int m = m0 + row, k = q * 4;
    float4 v = make_float4(0.0f, 0.0f, 0.0f, 0.0f);
    if (m < n && k < K) v = *reinterpret_cast<const float4*>(X + (long)m * K + k);
    ushort4 s;
    s.x = f2bf(v.x); s.y = f2bf(v.y); s.z = f2bf(v.z); s.w = f2bf(v.w);
    *reinterpret_cast<ushort4*>(&As[row][k]) = s;
  }
  __syncthreads();

  const int wave = tid >> 5;
  const int lane = tid & 31;
  const int half = lane >> 4;        // 0: lanes 0-15, 1: lanes 16-31
  const int l16  = lane & 15;
  const int col  = wave * 16 + l16;  // output column
  const unsigned short* wrow = Wt + (long)col * KP;

  v8f acc = {};
  #pragma unroll
  for (int k0 = 0; k0 < KP; k0 += 32) {
    // A 16x32 bf16 fragment: lane holds row l16,
    //   elems 0-7  = K k0+8*half   .. +7   (contiguous)
    //   elems 8-15 = K k0+16+8*half .. +7  (contiguous)
    v8u a_lo = *reinterpret_cast<const v8u*>(&As[l16][k0 + 8 * half]);
    v8u a_hi = *reinterpret_cast<const v8u*>(&As[l16][k0 + 16 + 8 * half]);
    v16u au  = __builtin_shufflevector(a_lo, a_hi,
                 0, 1, 2, 3, 4, 5, 6, 7, 8, 9, 10, 11, 12, 13, 14, 15);
    // B 32x16 bf16 fragment: lane = column, K = k0+8*half .. +7, then k0+16+8*half .. +7 (same as A)
    v8u b_lo = *reinterpret_cast<const v8u*>(wrow + k0 + 8 * half);
    v8u b_hi = *reinterpret_cast<const v8u*>(wrow + k0 + 16 + 8 * half);
    v16u bu  = __builtin_shufflevector(b_lo, b_hi,
                 0, 1, 2, 3, 4, 5, 6, 7, 8, 9, 10, 11, 12, 13, 14, 15);

    v16bf a = __builtin_bit_cast(v16bf, au);
    v16bf b = __builtin_bit_cast(v16bf, bu);
    acc = __builtin_amdgcn_wmma_f32_16x16x32_bf16(
        /*neg_a=*/false, a, /*neg_b=*/false, b,
        /*c_mod=*/(short)0, acc, /*reuse_a=*/false, /*reuse_b=*/false);
  }

  // C/D layout: VGPR j -> M = j + 8*half, N = lane&15
  #pragma unroll
  for (int j = 0; j < 8; ++j) {
    int m = m0 + j + half * 8;
    if (m < n) Y[(long)m * N + col] = acc[j];
  }
}

// ---------- aggregation ----------
template <int F>
__global__ void self_bias_init_k(const float* __restrict__ xw, const float* __restrict__ dinv,
                                 const float* __restrict__ bias, float* __restrict__ out, int n) {
  long t = (long)blockIdx.x * blockDim.x + threadIdx.x;
  long total = (long)n * F;
  if (t >= total) return;
  int i = (int)(t / F);
  int c = (int)(t % F);
  float di = dinv[i];
  out[t] = bias[c] + xw[t] * di * di;
}

template <int F>
__global__ void gcn_scatter_k(const float* __restrict__ xw, const int* __restrict__ src,
                              const int* __restrict__ dst, const float* __restrict__ dinv,
                              float* __restrict__ out, int E) {
  constexpr int G = F / 4;
  long tid = (long)blockIdx.x * blockDim.x + threadIdx.x;
  long total = (long)E * G;
  if (tid >= total) return;
  int e = (int)(tid / G);
  int g = (int)(tid % G);
  int s = src[e], d = dst[e];
  float nrm = dinv[s] * dinv[d];
  const float4 v = *reinterpret_cast<const float4*>(xw + (long)s * F + g * 4);
  float* o = out + (long)d * F + g * 4;
  atomicAdd(o + 0, v.x * nrm);
  atomicAdd(o + 1, v.y * nrm);
  atomicAdd(o + 2, v.z * nrm);
  atomicAdd(o + 3, v.w * nrm);
}

__global__ void relu_k(float* __restrict__ h, long total) {
  long t = (long)blockIdx.x * blockDim.x + threadIdx.x;
  if (t < total) h[t] = fmaxf(h[t], 0.0f);
}

// ---------- launch ----------
extern "C" void kernel_launch(void* const* d_in, const int* in_sizes, int n_in,
                              void* d_out, int out_size, void* d_ws, size_t ws_size,
                              hipStream_t stream) {
  const float* x  = (const float*)d_in[0];   // [n, 300]
  const int*   ei = (const int*)  d_in[1];   // [2, E]
  const float* W1 = (const float*)d_in[2];   // [300, 128]
  const float* b1 = (const float*)d_in[3];   // [128]
  const float* W2 = (const float*)d_in[4];   // [128, 64]
  const float* b2 = (const float*)d_in[5];   // [64]
  float* out = (float*)d_out;                // [n, 64]

  const int n = in_sizes[0] / 300;
  const int E = in_sizes[1] / 2;
  const int* srcv = ei;       // edge_index[0]
  const int* dstv = ei + E;   // edge_index[1]

  // workspace layout (256B aligned blocks)
  auto align256 = [](size_t s) { return (s + 255) & ~(size_t)255; };
  char* ws = (char*)d_ws;
  size_t off = 0;
  float* dinv = (float*)(ws + off);  off += align256((size_t)n * sizeof(float));
  float* xw   = (float*)(ws + off);  off += align256((size_t)n * 128 * sizeof(float));
  float* h1   = (float*)(ws + off);  off += align256((size_t)n * 128 * sizeof(float));
  unsigned short* Wt1 = (unsigned short*)(ws + off);  off += align256((size_t)128 * 320 * 2);
  unsigned short* Wt2 = (unsigned short*)(ws + off);  off += align256((size_t)64 * 128 * 2);
  (void)ws_size; (void)n_in; (void)out_size;

  const int B = 256;
  const int mt = (n + 15) / 16;

  // normalization coefficients
  deg_init_k  <<<(n + B - 1) / B, B, 0, stream>>>(dinv, n);
  deg_count_k <<<(E + B - 1) / B, B, 0, stream>>>(dstv, dinv, E);
  deg_finish_k<<<(n + B - 1) / B, B, 0, stream>>>(dinv, n);

  // weight pre-transpose / convert (bf16, zero-padded K)
  wcvt_k<<<(128 * 320 + B - 1) / B, B, 0, stream>>>(W1, Wt1, 300, 320, 128);
  wcvt_k<<<( 64 * 128 + B - 1) / B, B, 0, stream>>>(W2, Wt2, 128, 128,  64);

  // ----- layer 1: xw = x @ W1 ; h1 = agg(xw) + b1 ; relu -----
  gcn_gemm_bf16<300, 320, 128><<<mt, 256, 0, stream>>>(x, Wt1, xw, n);
  {
    long tot = (long)n * 128;
    self_bias_init_k<128><<<(unsigned)((tot + B - 1) / B), B, 0, stream>>>(xw, dinv, b1, h1, n);
  }
  {
    long tot = (long)E * 32;
    gcn_scatter_k<128><<<(unsigned)((tot + B - 1) / B), B, 0, stream>>>(xw, srcv, dstv, dinv, h1, E);
  }
  {
    long tot = (long)n * 128;
    relu_k<<<(unsigned)((tot + B - 1) / B), B, 0, stream>>>(h1, tot);
  }

  // ----- layer 2: xw = h1 @ W2 ; out = agg(xw) + b2 -----
  gcn_gemm_bf16<128, 128, 64><<<mt, 128, 0, stream>>>(h1, Wt2, xw, n);
  {
    long tot = (long)n * 64;
    self_bias_init_k<64><<<(unsigned)((tot + B - 1) / B), B, 0, stream>>>(xw, dinv, b2, out, n);
  }
  {
    long tot = (long)E * 16;
    gcn_scatter_k<64><<<(unsigned)((tot + B - 1) / B), B, 0, stream>>>(xw, srcv, dstv, dinv, out, E);
  }
}
